// MPNNWithAttention_26027501813983
// MI455X (gfx1250) — matrix operand order repaired and checked by reading, not compiled, not run on hardware
//
#include <hip/hip_runtime.h>
#include <hip/hip_bf16.h>

#define N_NODES 50000
#define E_EDGES 500000
#define EP      (E_EDGES + N_NODES)   // edges + self loops
#define IN_F    128
#define HID     64
#define HEADS   4
#define G_GR    16
#define OUT_F   326000
#define NT_FC   (OUT_F / 16)          // 20375 column tiles in the FC
#define NEG_SLOPE 0.2f

typedef _Float16 half16 __attribute__((ext_vector_type(16)));
typedef float    float8 __attribute__((ext_vector_type(8)));

// ---------------- WMMA helpers (CDNA5 16x16x32 f16 -> f32) ----------------

static __device__ __forceinline__ float8 wmma_f16(half16 a, half16 b, float8 c) {
  // (neg_a, A, neg_b, B, c_mod, C, reuse_a, reuse_b)
  return __builtin_amdgcn_wmma_f32_16x16x32_f16(false, a, false, b, (short)0, c,
                                                false, false);
}

// A tile 16x32 (MxK) from an f16 LDS strip [16, ld]. Lane l: M = l&15,
// elems 0..15 -> K = kh+0..15, kh = (l>>4)*16.
// (contiguous k order, not the ISA table order; load_b_tile matches it)
static __device__ __forceinline__ half16 load_a_lds(const _Float16* As, int ld,
                                                    int k0, int lane) {
  const _Float16* row = As + (lane & 15) * ld + k0 + ((lane >> 4) * 16);
  half16 a;
#pragma unroll
  for (int i = 0; i < 16; ++i) a[i] = row[i];
  return a;
}

// B tile 32x16 (KxN), f32 source row-major [K, ld]. Lane l: N = l&15,
// lanes 0-15 hold K = k0+0..15, lanes 16-31 hold K = k0+16..31.
static __device__ __forceinline__ half16 load_b_tile(const float* W, int ld,
                                                     int k0, int n0, int lane) {
  const int n  = n0 + (lane & 15);
  const int ks = k0 + (lane >> 4) * 16;
  half16 b;
#pragma unroll
  for (int i = 0; i < 16; ++i) b[i] = (_Float16)W[(long)(ks + i) * ld + n];
  return b;
}

// ---------------- monotonic float<->uint for atomic segment-max ----------------
static __device__ __forceinline__ unsigned encf(float f) {
  unsigned b = __float_as_uint(f);
  return (b & 0x80000000u) ? ~b : (b | 0x80000000u);
}
static __device__ __forceinline__ float decf(unsigned u) {
  unsigned b = (u & 0x80000000u) ? (u & 0x7FFFFFFFu) : ~u;
  return __uint_as_float(b);
}

static __device__ __forceinline__ void edge_sd(const int* eidx, int e, int& s, int& d) {
  if (e < E_EDGES) { s = eidx[e]; d = eidx[E_EDGES + e]; }
  else             { s = d = e - E_EDGES; }           // self loop
}

// ---------------- kernels ----------------

// h = x @ W_emb + b_emb   [N,128]x[128,64]
// 128-thread block = 4 waves share one 16-row A strip (staged f16 in LDS);
// wave w computes column tile n0 = w*16 (HID = 4 tiles).
__global__ void k_embed(const float* __restrict__ x, const float* __restrict__ W,
                        const float* __restrict__ bias, float* __restrict__ h) {
  __shared__ _Float16 As[16 * (IN_F + 8)];
  const int tid = threadIdx.x;
  const int m0  = blockIdx.x * 16;
  for (int i = tid; i < 16 * IN_F; i += 128) {
    const int m = i >> 7, k = i & (IN_F - 1);
    As[m * (IN_F + 8) + k] = (_Float16)x[(long)(m0 + m) * IN_F + k];
  }
  __syncthreads();
  const int lane = tid & 31;
  const int n0   = (tid >> 5) * 16;
  float8 c = {};
  for (int k0 = 0; k0 < IN_F; k0 += 32) {
    half16 a = load_a_lds(As, IN_F + 8, k0, lane);
    half16 b = load_b_tile(W, HID, k0, n0, lane);
    c = wmma_f16(a, b, c);
  }
  const int n  = n0 + (lane & 15);
  const int mb = (lane >> 4) * 8;
  const float bv = bias[n];
#pragma unroll
  for (int r = 0; r < 8; ++r) h[(m0 + mb + r) * HID + n] = c[r] + bv;
}

// z = h @ W_gat[l]   [N,64]x[64,256]
// 128-thread block: shared 16-row f16 A strip in LDS; wave w computes
// 4 column tiles n0 = w*64 + t*16 (8 WMMAs/wave, A fragment reused 4x).
__global__ void k_zgemm(const float* __restrict__ h, const float* __restrict__ W,
                        float* __restrict__ z) {
  __shared__ _Float16 As[16 * (HID + 8)];
  const int tid = threadIdx.x;
  const int m0  = blockIdx.x * 16;
  for (int i = tid; i < 16 * HID; i += 128) {
    const int m = i >> 6, k = i & (HID - 1);
    As[m * (HID + 8) + k] = (_Float16)h[(long)(m0 + m) * HID + k];
  }
  __syncthreads();
  const int lane = tid & 31;
  const int wv   = tid >> 5;
  float8 c[4] = {};
  for (int k0 = 0; k0 < HID; k0 += 32) {
    half16 a = load_a_lds(As, HID + 8, k0, lane);
#pragma unroll
    for (int t = 0; t < 4; ++t) {
      half16 b = load_b_tile(W, HEADS * HID, k0, wv * 64 + t * 16, lane);
      c[t] = wmma_f16(a, b, c[t]);
    }
  }
  const int mb = (lane >> 4) * 8;
#pragma unroll
  for (int t = 0; t < 4; ++t) {
    const int n = wv * 64 + t * 16 + (lane & 15);
#pragma unroll
    for (int r = 0; r < 8; ++r)
      z[(long)(m0 + mb + r) * (HEADS * HID) + n] = c[t][r];
  }
}

// al_s[n,h] = <z[n,h,:], a_src[h,:]>, same for a_dst
__global__ void k_att(const float* __restrict__ z, const float* __restrict__ asrc,
                      const float* __restrict__ adst, float* __restrict__ als,
                      float* __restrict__ ald) {
  const int idx = blockIdx.x * blockDim.x + threadIdx.x;
  if (idx >= N_NODES * HEADS) return;
  const int n = idx >> 2, h = idx & 3;
  const float* zr = z + (long)n * (HEADS * HID) + h * HID;
  const float* as = asrc + h * HID;
  const float* ad = adst + h * HID;
  float ss = 0.f, sd = 0.f;
#pragma unroll 8
  for (int f = 0; f < HID; ++f) { float v = zr[f]; ss += v * as[f]; sd += v * ad[f]; }
  als[idx] = ss; ald[idx] = sd;
}

// pass 1: segment max of leaky_relu(al_s[src]+al_d[dst]) over dst
__global__ void k_edge_max(const int* __restrict__ eidx, const float* __restrict__ als,
                           const float* __restrict__ ald, unsigned* __restrict__ emax) {
  const int idx = blockIdx.x * blockDim.x + threadIdx.x;
  if (idx >= EP * HEADS) return;
  const int e = idx >> 2, h = idx & 3;
  int s, d; edge_sd(eidx, e, s, d);
  float ev = als[s * HEADS + h] + ald[d * HEADS + h];
  ev = (ev >= 0.f) ? ev : NEG_SLOPE * ev;
  atomicMax(&emax[d * HEADS + h], encf(ev));
}

// pass 2: w = exp(e - max), denom = segment_sum(w)
__global__ void k_edge_w(const int* __restrict__ eidx, const float* __restrict__ als,
                         const float* __restrict__ ald, const unsigned* __restrict__ emax,
                         float* __restrict__ wbuf, float* __restrict__ denom) {
  const int idx = blockIdx.x * blockDim.x + threadIdx.x;
  if (idx >= EP * HEADS) return;
  const int e = idx >> 2, h = idx & 3;
  int s, d; edge_sd(eidx, e, s, d);
  float ev = als[s * HEADS + h] + ald[d * HEADS + h];
  ev = (ev >= 0.f) ? ev : NEG_SLOPE * ev;
  const float w = __expf(ev - decf(emax[d * HEADS + h]));
  wbuf[idx] = w;
  atomicAdd(&denom[d * HEADS + h], w);
}

// pass 3: acc[dst,f] += sum_h alpha[e,h]*z[src,h,f]  (head-mean folded later)
// one wave per edge, 4 waves per block
__global__ void k_aggregate(const int* __restrict__ eidx, const float* __restrict__ z,
                            const float* __restrict__ wbuf, const float* __restrict__ denom,
                            float* __restrict__ acc) {
  const int e = blockIdx.x * 4 + (threadIdx.x >> 5);
  if (e >= EP) return;
  const int lane = threadIdx.x & 31;
  int s, d; edge_sd(eidx, e, s, d);
  float alpha[HEADS];
#pragma unroll
  for (int h = 0; h < HEADS; ++h)
    alpha[h] = wbuf[e * HEADS + h] / (denom[d * HEADS + h] + 1e-16f);
  const float* zs = z + (long)s * (HEADS * HID);
  float* ad = acc + (long)d * HID;
#pragma unroll
  for (int j = 0; j < 2; ++j) {
    const int f = lane + j * 32;
    float m = 0.f;
#pragma unroll
    for (int h = 0; h < HEADS; ++h) m += alpha[h] * zs[h * HID + f];
    atomicAdd(&ad[f], m);
  }
}

// h_next = relu(acc/HEADS + b)   (in place)
__global__ void k_finalize(float* __restrict__ h, const float* __restrict__ bias) {
  const int idx = blockIdx.x * blockDim.x + threadIdx.x;
  if (idx >= N_NODES * HID) return;
  const float v = h[idx] * (1.f / HEADS) + bias[idx & (HID - 1)];
  h[idx] = v > 0.f ? v : 0.f;
}

__global__ void k_pool_acc(const float* __restrict__ h, const int* __restrict__ batch,
                           float* __restrict__ sums) {
  const int idx = blockIdx.x * blockDim.x + threadIdx.x;
  if (idx >= N_NODES * HID) return;
  const int n = idx >> 6, f = idx & (HID - 1);
  atomicAdd(&sums[batch[n] * HID + f], h[idx]);
}

__global__ void k_pool_cnt(const int* __restrict__ batch, float* __restrict__ cnts) {
  const int n = blockIdx.x * blockDim.x + threadIdx.x;
  if (n >= N_NODES) return;
  atomicAdd(&cnts[batch[n]], 1.f);
}

__global__ void k_pool_fin(float* __restrict__ sums, const float* __restrict__ cnts) {
  const int idx = blockIdx.x * blockDim.x + threadIdx.x;
  if (idx >= G_GR * HID) return;
  sums[idx] = sums[idx] / fmaxf(cnts[idx >> 6], 1.f);
}

// out = hg @ W_fc + b_fc   [16,64]x[64,326000]; G==16 -> one M-tile.
// 128-thread block = 4 waves; hg staged once in LDS as f16; one n-tile per
// wave; prefetch the B stream one column-group ahead (global_prefetch_b8).
__global__ void k_fc(const float* __restrict__ hg, const float* __restrict__ W,
                     const float* __restrict__ bias, float* __restrict__ out) {
  __shared__ _Float16 As[16 * (HID + 8)];
  const int tid = threadIdx.x;
  for (int i = tid; i < 16 * HID; i += 128) {
    const int m = i >> 6, k = i & (HID - 1);
    As[m * (HID + 8) + k] = (_Float16)hg[m * HID + k];
  }
  __syncthreads();
  const int lane = tid & 31;
  const int tile = blockIdx.x * 4 + (tid >> 5);
  if (tile >= NT_FC) return;            // wave-uniform exit (EXEC stays full)
  const int n0 = tile * 16;
  // prefetch next column-group of the B stream (rows lane and lane+32)
  if (n0 + 64 < OUT_F) {
    __builtin_prefetch(&W[(long)lane * OUT_F + n0 + 64], 0, 1);
    __builtin_prefetch(&W[(long)(lane + 32) * OUT_F + n0 + 64], 0, 1);
  }
  float8 c = {};
  for (int k0 = 0; k0 < HID; k0 += 32) {
    half16 a = load_a_lds(As, HID + 8, k0, lane);
    half16 b = load_b_tile(W, OUT_F, k0, n0, lane);
    c = wmma_f16(a, b, c);
  }
  const int n  = n0 + (lane & 15);
  const int mb = (lane >> 4) * 8;
  const float bv = bias[n];
#pragma unroll
  for (int r = 0; r < 8; ++r) out[(long)(mb + r) * OUT_F + n] = c[r] + bv;
}

// ---------------- launch ----------------

extern "C" void kernel_launch(void* const* d_in, const int* in_sizes, int n_in,
                              void* d_out, int out_size, void* d_ws, size_t ws_size,
                              hipStream_t stream) {
  (void)in_sizes; (void)n_in; (void)out_size; (void)ws_size;
  const float* x       = (const float*)d_in[0];
  const int*   eidx    = (const int*)d_in[1];
  // d_in[2] = edge_attr (unused by the reference forward)
  const int*   batch   = (const int*)d_in[3];
  const float* W_emb   = (const float*)d_in[4];
  const float* b_emb   = (const float*)d_in[5];
  const float* W_gat   = (const float*)d_in[6];
  const float* att_src = (const float*)d_in[7];
  const float* att_dst = (const float*)d_in[8];
  const float* b_gat   = (const float*)d_in[9];
  const float* W_fc    = (const float*)d_in[10];
  const float* b_fc    = (const float*)d_in[11];
  float* out = (float*)d_out;

  char* ws = (char*)d_ws;
  size_t off = 0;
  auto alloc = [&](size_t bytes) -> void* {
    void* p = (void*)(ws + off);
    off += (bytes + 255) & ~(size_t)255;
    return p;
  };
  float*    hA    = (float*)alloc((size_t)N_NODES * HID * 4);
  float*    hB    = (float*)alloc((size_t)N_NODES * HID * 4);
  float*    z     = (float*)alloc((size_t)N_NODES * HEADS * HID * 4);
  float*    als   = (float*)alloc((size_t)N_NODES * HEADS * 4);
  float*    ald   = (float*)alloc((size_t)N_NODES * HEADS * 4);
  unsigned* emax  = (unsigned*)alloc((size_t)N_NODES * HEADS * 4);
  float*    denom = (float*)alloc((size_t)N_NODES * HEADS * 4);
  float*    wbuf  = (float*)alloc((size_t)EP * HEADS * 4);
  float*    sums  = (float*)alloc((size_t)G_GR * HID * 4);
  float*    cnts  = (float*)alloc((size_t)G_GR * 4);

  // node embedding
  k_embed<<<N_NODES / 16, 128, 0, stream>>>(x, W_emb, b_emb, hA);

  float* hcur = hA;
  float* hnxt = hB;
  for (int l = 0; l < 3; ++l) {
    k_zgemm<<<N_NODES / 16, 128, 0, stream>>>(
        hcur, W_gat + (size_t)l * HID * HEADS * HID, z);
    k_att<<<(N_NODES * HEADS + 255) / 256, 256, 0, stream>>>(
        z, att_src + (size_t)l * HEADS * HID, att_dst + (size_t)l * HEADS * HID,
        als, ald);
    hipMemsetAsync(emax, 0, (size_t)N_NODES * HEADS * 4, stream);
    hipMemsetAsync(denom, 0, (size_t)N_NODES * HEADS * 4, stream);
    hipMemsetAsync(hnxt, 0, (size_t)N_NODES * HID * 4, stream);
    k_edge_max<<<(EP * HEADS + 255) / 256, 256, 0, stream>>>(eidx, als, ald, emax);
    k_edge_w<<<(EP * HEADS + 255) / 256, 256, 0, stream>>>(eidx, als, ald, emax,
                                                           wbuf, denom);
    k_aggregate<<<(EP + 3) / 4, 128, 0, stream>>>(eidx, z, wbuf, denom, hnxt);
    k_finalize<<<(N_NODES * HID + 255) / 256, 256, 0, stream>>>(hnxt,
                                                                b_gat + (size_t)l * HID);
    float* t = hcur; hcur = hnxt; hnxt = t;
  }

  // global mean pool
  hipMemsetAsync(sums, 0, (size_t)G_GR * HID * 4, stream);
  hipMemsetAsync(cnts, 0, (size_t)G_GR * 4, stream);
  k_pool_acc<<<(N_NODES * HID + 255) / 256, 256, 0, stream>>>(hcur, batch, sums);
  k_pool_cnt<<<(N_NODES + 255) / 256, 256, 0, stream>>>(batch, cnts);
  k_pool_fin<<<(G_GR * HID + 255) / 256, 256, 0, stream>>>(sums, cnts);

  // final FC
  k_fc<<<(NT_FC + 3) / 4, 128, 0, stream>>>(sums, W_fc, b_fc, out);
}
